// Scale_Dot_Product_Attention_18193481466322
// MI455X (gfx1250) — matrix operand order repaired and checked by reading, not compiled, not run on hardware
//
#include <hip/hip_runtime.h>
#include <hip/hip_bf16.h>
#include <math.h>

typedef __attribute__((ext_vector_type(16))) __bf16 v16bf;
typedef __attribute__((ext_vector_type(8)))  float  v8f;

#define B_  2
#define H_  16
#define S_  2048
#define D_  64
#define QT  32            // query rows per block
#define NTHREADS 256      // 8 wave32s
#define STRIDE 2052       // padded LDS score-row stride (floats)
#define CHUNK 128         // key rows staged per iteration
#define NCHUNK (S_ / CHUNK)   // 16
#define KSTRIDE 72        // bf16 per staged K row   (64 + 8 pad, 144 B: 16B-aligned)
#define VSTRIDE 136       // bf16 per staged Vt row (128 + 8 pad, 272 B: 16B-aligned)
#define SCALE 0.125f      // 1/sqrt(64)

#define SC_FLOATS    (QT * STRIDE)          // 65664 floats = 262656 B
#define STAGE_FLOATS 4608                   // 18432 B  (K: 128*72*2; Vt: 64*136*2)
#define RED_OFF      (SC_FLOATS + STAGE_FLOATS)
#define SMEM_FLOATS  (RED_OFF + NTHREADS + QT + QT)

// 16 contiguous bf16 (32 B, 16B-aligned) -> A/B fragment half-lane payload
__device__ __forceinline__ v16bf load16bf(const __bf16* p) {
  struct U { uint4 a, b; } u;
  u.a = *(const uint4*)(p);
  u.b = *(const uint4*)(p + 8);
  return __builtin_bit_cast(v16bf, u);
}

// two 8-float contiguous runs (f32 scores in LDS) -> bf16 fragment
__device__ __forceinline__ v16bf cvt16(const float* lo, const float* hi) {
  v16bf r;
#pragma unroll
  for (int i = 0; i < 8; ++i) r[i] = (__bf16)lo[i];
#pragma unroll
  for (int i = 0; i < 8; ++i) r[8 + i] = (__bf16)hi[i];
  return r;
}

__global__ __launch_bounds__(NTHREADS, 1)
void sdpa_wmma_kernel(const float* __restrict__ q, const float* __restrict__ k,
                      const float* __restrict__ v, float* __restrict__ out,
                      float* __restrict__ attn) {
  extern __shared__ float smem[];
  float*  sc     = smem;                       // QT x STRIDE f32 scores
  __bf16* stage  = (__bf16*)(smem + SC_FLOATS);// K tile / transposed V tile (bf16)
  float*  red    = smem + RED_OFF;             // 256 partials
  float*  rowmax = red + NTHREADS;             // 32
  float*  rowinv = rowmax + QT;                // 32

  const int tid  = threadIdx.x;
  const int lane = tid & 31;
  const int wave = tid >> 5;
  const int nlo  = lane & 15;   // A-row / B-col / D-col
  const int hiK  = lane >> 4;   // lane half -> K sub-range
  const int rt   = wave >> 2;   // query row-tile (0/1)
  const int wq   = wave & 3;    // key-tile / out-col-tile slot

  const int bh = blockIdx.x >> 6;           // 64 q-tiles of 32 per (b,h)
  const int q0 = (blockIdx.x & 63) * QT;

  // ---- Q fragments for this wave's 16-row tile (loaded once) ----
  v16bf qa0, qa1;
  {
    const float* qp = q + ((size_t)bh * S_ + q0 + rt * 16 + nlo) * D_;
#pragma unroll
    for (int e = 0; e < 16; ++e) {
      int kk = e + hiK * 16;   // A-frag element -> k
      qa0[e] = (__bf16)qp[kk];
      qa1[e] = (__bf16)qp[32 + kk];
    }
  }

  // ================= Phase 1: S = scale * Q K^T =================
  for (int c = 0; c < NCHUNK; ++c) {
    __syncthreads();
    // cooperatively stage 128 key rows as bf16 (row-major, padded)
    const float* kg = k + ((size_t)bh * S_ + (size_t)c * CHUNK) * D_;
    for (int i = tid; i < CHUNK * D_ / 4; i += NTHREADS) {
      float4 f = ((const float4*)kg)[i];
      int row = i >> 4, col = (i & 15) * 4;
      __bf16* d = stage + row * KSTRIDE + col;
      d[0] = (__bf16)f.x; d[1] = (__bf16)f.y;
      d[2] = (__bf16)f.z; d[3] = (__bf16)f.w;
    }
    if (c + 1 < NCHUNK)   // pull next 32KB chunk toward the WGP (global_prefetch_b8)
      __builtin_prefetch((const char*)(kg + CHUNK * D_) + tid * 128, 0, 1);
    __syncthreads();
    // each wave: 2 of the 8 key tiles in this chunk (shared LDS K frags)
#pragma unroll
    for (int tt = 0; tt < 2; ++tt) {
      const int t = wq + tt * 4;
      const __bf16* kr = stage + (t * 16 + nlo) * KSTRIDE;  // key row = B column
      v16bf kb0 = load16bf(kr + hiK * 16);        // d = e + 16*hiK
      v16bf kb1 = load16bf(kr + 32 + hiK * 16);   // d = 32 + e + 16*hiK
      v8f acc = {};
      acc = __builtin_amdgcn_wmma_f32_16x16x32_bf16(false, qa0, false, kb0,
                                                    (short)0, acc, false, false);
      acc = __builtin_amdgcn_wmma_f32_16x16x32_bf16(false, qa1, false, kb1,
                                                    (short)0, acc, false, false);
      float* scol = sc + (c * 8 + t) * 16 + nlo;
#pragma unroll
      for (int r = 0; r < 8; ++r)
        scol[(rt * 16 + r + hiK * 8) * STRIDE] = acc[r] * SCALE;
    }
  }
  __syncthreads();

  // ================= Phase 2: row softmax + stream attn =================
  {
    const int row = tid >> 3, sub = tid & 7;   // 8 threads per row, 32 rows
    float* srow = sc + row * STRIDE;
    float m = -1e30f;
    for (int c2 = sub; c2 < S_; c2 += 8) m = fmaxf(m, srow[c2]);
    red[tid] = m;
    __syncthreads();
    if (sub == 0) {
      float mm = red[row * 8];
#pragma unroll
      for (int i = 1; i < 8; ++i) mm = fmaxf(mm, red[row * 8 + i]);
      rowmax[row] = mm;
    }
    __syncthreads();
    const float mm = rowmax[row];
    float s = 0.f;
    for (int c2 = sub; c2 < S_; c2 += 8) {
      float p = __expf(srow[c2] - mm);
      srow[c2] = p;
      s += p;
    }
    red[tid] = s;
    __syncthreads();
    if (sub == 0) {
      float ss = 0.f;
#pragma unroll
      for (int i = 0; i < 8; ++i) ss += red[row * 8 + i];
      rowinv[row] = 1.0f / ss;
    }
    __syncthreads();
    // normalize in LDS and stream attention matrix (256-wide coalesced stores)
    for (int r = 0; r < QT; ++r) {
      const float inv = rowinv[r];
      float* sr = sc + r * STRIDE;
      float* ar = attn + ((size_t)bh * S_ + q0 + r) * S_;
      for (int c2 = tid; c2 < S_; c2 += NTHREADS) {
        float p = sr[c2] * inv;
        sr[c2] = p;
        ar[c2] = p;
      }
    }
  }

  // ================= Phase 3: O = P V =================
  // wave (rt, wq): output rows [rt*16, rt*16+16), cols [wq*16, wq*16+16)
  {
    const int d0 = wq * 16 + nlo;
    v8f acc = {};
    for (int c = 0; c < NCHUNK; ++c) {
      __syncthreads();   // also covers phase-2 sc writes on first iteration
      // stage 128 V rows TRANSPOSED (d-major) as bf16 so B-frags are contiguous
      const float* vg = v + ((size_t)bh * S_ + (size_t)c * CHUNK) * D_;
      for (int i = tid; i < CHUNK * D_ / 4; i += NTHREADS) {
        float4 f = ((const float4*)vg)[i];
        int krow = i >> 4, col = (i & 15) * 4;
        stage[(col + 0) * VSTRIDE + krow] = (__bf16)f.x;
        stage[(col + 1) * VSTRIDE + krow] = (__bf16)f.y;
        stage[(col + 2) * VSTRIDE + krow] = (__bf16)f.z;
        stage[(col + 3) * VSTRIDE + krow] = (__bf16)f.w;
      }
      if (c + 1 < NCHUNK)
        __builtin_prefetch((const char*)(vg + CHUNK * D_) + tid * 128, 0, 1);
      __syncthreads();
      const __bf16* vrow = stage + d0 * VSTRIDE;                  // this lane's out col
      const float*  srow = sc + (rt * 16 + nlo) * STRIDE + c * CHUNK;
#pragma unroll
      for (int t = 0; t < 4; ++t) {   // 4 K=32 sub-tiles per chunk
        v16bf pa = cvt16(srow + t * 32 + hiK * 16,
                         srow + t * 32 + 8 + hiK * 16);
        v16bf vb = load16bf(vrow + t * 32 + hiK * 16);
        acc = __builtin_amdgcn_wmma_f32_16x16x32_bf16(false, pa, false, vb,
                                                      (short)0, acc, false, false);
      }
    }
#pragma unroll
    for (int r = 0; r < 8; ++r)
      out[((size_t)bh * S_ + q0 + rt * 16 + r + hiK * 8) * D_ + d0] = acc[r];
  }
}

extern "C" void kernel_launch(void* const* d_in, const int* in_sizes, int n_in,
                              void* d_out, int out_size, void* d_ws, size_t ws_size,
                              hipStream_t stream) {
  const float* q = (const float*)d_in[0];
  const float* k = (const float*)d_in[1];
  const float* v = (const float*)d_in[2];
  float* out  = (float*)d_out;                       // [B,H,S,D]
  float* attn = out + (size_t)B_ * H_ * S_ * D_;     // [B,H,S,S]

  const size_t smem_bytes = (size_t)SMEM_FLOATS * sizeof(float);  // ~282 KB
  (void)hipFuncSetAttribute((const void*)sdpa_wmma_kernel,
                            hipFuncAttributeMaxDynamicSharedMemorySize,
                            (int)smem_bytes);

  const int nblocks = B_ * H_ * (S_ / QT);   // 2048
  sdpa_wmma_kernel<<<nblocks, NTHREADS, smem_bytes, stream>>>(q, k, v, out, attn);
}
